// GATModel_49959059587121
// MI455X (gfx1250) — matrix operand order repaired and checked by reading, not compiled, not run on hardware
//
#include <hip/hip_runtime.h>
#include <hip/hip_bf16.h>

// ---------------- problem constants ----------------
#define NN   50000      // nodes
#define EE   800000     // edges (before self loops)
#define ETOT (EE + NN)  // edges incl. self loops
#define FIN  128        // in_channels
#define HH   8          // heads
#define CC   32         // channels per head
#define HC   256        // H*C
#define GG   256        // graphs
#define NEG_SLOPE 0.2f

typedef __attribute__((ext_vector_type(16))) _Float16 v16h;
typedef __attribute__((ext_vector_type(8)))  float    v8f;

// ---------------- helpers ----------------
__device__ __forceinline__ void atomicMaxF(float* addr, float v) {
    // classic float-max via integer atomics (native global_atomic_max/min)
    if (v >= 0.0f) atomicMax((int*)addr, __float_as_int(v));
    else           atomicMin((unsigned int*)addr, __float_as_uint(v));
}

__device__ __forceinline__ float lrelu(float v) {
    return v > 0.0f ? v : NEG_SLOPE * v;
}

// ---------------- fill ----------------
__global__ void fill_u32(unsigned int* p, int n, unsigned int val) {
    int i = blockIdx.x * blockDim.x + threadIdx.x;
    if (i < n) p[i] = val;
}

// ---------------- GEMM: h[n,o] = sum_k X[n,k] * W[o,k] ----------------
// One wave computes a 16x32 strip of h (two 16x16 WMMA tiles sharing one A
// fragment) using v_wmma_f32_16x16x32_f16 with f32 accumulation.
// K is a template parameter (128 or 256) so the K loop fully unrolls:
// all loads for the strip issue up front, WMMAs pipeline behind them.
// Grid: (N/16)*(HC/32)/8 = 3125 blocks x 8 waves -- exact division, so EXEC
// is all-ones for every WMMA (ISA requirement).
template <int K>
__global__ void gemm_wmma(const float* __restrict__ X,   // [N, K]
                          const float* __restrict__ W,   // [HC, K]
                          float* __restrict__ Hout) {    // [N, HC]
    const int lane = threadIdx.x & 31;
    const int wv   = threadIdx.x >> 5;
    const int t    = blockIdx.x * 8 + wv;      // global strip id, 0..24999
    const int tm   = t >> 3;                   // M tile (16 rows), 0..3124
    const int tn   = t & 7;                    // N strip (32 cols = 1 head)
    const int l15  = lane & 15;
    const int hsel = lane >> 4;

    const int arow  = tm * 16 + l15;           // A row this lane feeds
    const int bcol0 = tn * 32 + l15;           // B column, tile 0
    const int bcol1 = tn * 32 + 16 + l15;      // B column, tile 1

    v8f c0 = {}, c1 = {};
#pragma unroll
    for (int k0 = 0; k0 < K; k0 += 32) {
        // A lane layout (16-bit A 16x32, ISA 7.12.2):
        //  lanes 0-15 : K = {k0..k0+7, k0+16..k0+23}
        //  lanes 16-31: K = {k0+8..k0+15, k0+24..k0+31}
        v16h a, b0, b1;
        const float* xr = X + (size_t)arow * K + k0 + hsel * 8;
#pragma unroll
        for (int i = 0; i < 8; ++i)  a[i] = (_Float16)xr[i];
#pragma unroll
        for (int i = 8; i < 16; ++i) a[i] = (_Float16)xr[i + 8];
        // B lane layout (32x16): same K order as A:
        //  lanes 0-15 : K = {k0..k0+7, k0+16..k0+23}; lanes 16-31: K = {k0+8..k0+15, k0+24..k0+31}
        const float* wr0 = W + (size_t)bcol0 * K + k0 + hsel * 8;
        const float* wr1 = W + (size_t)bcol1 * K + k0 + hsel * 8;
#pragma unroll
        for (int i = 0; i < 16; ++i) { b0[i] = (_Float16)wr0[i < 8 ? i : i + 8];
                                       b1[i] = (_Float16)wr1[i < 8 ? i : i + 8]; }

        c0 = __builtin_amdgcn_wmma_f32_16x16x32_f16(
                 false, a, false, b0, (short)0, c0, false, false);
        c1 = __builtin_amdgcn_wmma_f32_16x16x32_f16(
                 false, a, false, b1, (short)0, c1, false, false);
    }
    // C/D layout: VGPR r -> row (tm*16 + 8*hsel + r), col = base + (lane&15)
    float* hp = Hout + (size_t)(tm * 16 + hsel * 8) * HC + tn * 32 + l15;
#pragma unroll
    for (int r = 0; r < 8; ++r) {
        hp[(size_t)r * HC]      = c0[r];
        hp[(size_t)r * HC + 16] = c1[r];
    }
}

// ---------------- attention coefficients a_s, a_d ----------------
__global__ void attn_dots(const float* __restrict__ Hm,
                          const float* __restrict__ att_s,
                          const float* __restrict__ att_d,
                          float* __restrict__ as_out,
                          float* __restrict__ ad_out) {
    int idx = blockIdx.x * blockDim.x + threadIdx.x;    // n*H + h
    if (idx >= NN * HH) return;
    int n = idx >> 3, h = idx & 7;
    const float* hp = Hm + (size_t)n * HC + h * CC;
    const float* sp = att_s + h * CC;
    const float* dp = att_d + h * CC;
    float s = 0.f, d = 0.f;
#pragma unroll
    for (int c = 0; c < CC; ++c) { s += hp[c] * sp[c]; d += hp[c] * dp[c]; }
    as_out[idx] = s;
    ad_out[idx] = d;
}

// ---------------- edge pass 1: segment max ----------------
__global__ void edge_max(const int* __restrict__ ei,
                         const float* __restrict__ as_,
                         const float* __restrict__ ad_,
                         float* __restrict__ m) {
    int e = blockIdx.x * blockDim.x + threadIdx.x;
    if (e >= ETOT) return;
    int s, d;
    if (e < EE) { s = ei[e]; d = ei[EE + e]; } else { s = d = e - EE; }
#pragma unroll
    for (int h = 0; h < HH; ++h) {
        float v = lrelu(as_[s * HH + h] + ad_[d * HH + h]);
        atomicMaxF(&m[d * HH + h], v);
    }
}

// ---------------- edge pass 2: exp + segment sum ----------------
__global__ void edge_expsum(const int* __restrict__ ei,
                            const float* __restrict__ as_,
                            const float* __restrict__ ad_,
                            const float* __restrict__ m,
                            float* __restrict__ denom) {
    int e = blockIdx.x * blockDim.x + threadIdx.x;
    if (e >= ETOT) return;
    int s, d;
    if (e < EE) { s = ei[e]; d = ei[EE + e]; } else { s = d = e - EE; }
#pragma unroll
    for (int h = 0; h < HH; ++h) {
        float v = lrelu(as_[s * HH + h] + ad_[d * HH + h]);
        float p = __expf(v - m[d * HH + h]);
        atomicAdd(&denom[d * HH + h], p);
    }
}

// ---------------- edge pass 3: alpha * h[src] scatter-add ----------------
// One wave (32 lanes) per edge: coalesced 128B gathers of h[src] (L2-hot)
// and coalesced float atomic adds into out[dst].
__global__ void edge_scatter(const int* __restrict__ ei,
                             const float* __restrict__ as_,
                             const float* __restrict__ ad_,
                             const float* __restrict__ m,
                             const float* __restrict__ denom,
                             const float* __restrict__ Hm,
                             float* __restrict__ Out) {
    int wv = threadIdx.x >> 5;
    int e  = blockIdx.x * 8 + wv;
    if (e >= ETOT) return;                       // wave-uniform exit
    int lane = threadIdx.x & 31;
    int s, d;
    if (e < EE) { s = ei[e]; d = ei[EE + e]; } else { s = d = e - EE; }

    float alpha_mine = 0.f;
    if (lane < HH) {
        float v = lrelu(as_[s * HH + lane] + ad_[d * HH + lane]);
        alpha_mine = __expf(v - m[d * HH + lane]) / denom[d * HH + lane];
    }
    const float* hp = Hm  + (size_t)s * HC;
    float*       op = Out + (size_t)d * HC;
#pragma unroll
    for (int j = 0; j < HH; ++j) {
        float aj = __shfl(alpha_mine, j, 32);
        float hv = hp[j * CC + lane];
        atomicAdd(&op[j * CC + lane], aj * hv);
    }
}

// ---------------- bias (+ optional relu) ----------------
__global__ void finalize_bias(float* __restrict__ Out,
                              const float* __restrict__ bias,
                              int do_relu) {
    int idx = blockIdx.x * blockDim.x + threadIdx.x;
    if (idx >= NN * HC) return;
    float v = Out[idx] + bias[idx & (HC - 1)];
    if (do_relu) v = v > 0.f ? v : 0.f;
    Out[idx] = v;
}

// ---------------- global mean pool (sums + counts) ----------------
__global__ void pool_sum(const float* __restrict__ X,
                         const int* __restrict__ batch,
                         float* __restrict__ pooled,
                         float* __restrict__ cnt) {
    int n = blockIdx.x, t = threadIdx.x;
    int g = batch[n];
    atomicAdd(&pooled[(size_t)g * HC + t], X[(size_t)n * HC + t]);
    if (t == 0) atomicAdd(&cnt[g], 1.0f);
}

// ---------------- final linear [G,HC] x [HC] -> [G,1] ----------------
__global__ void final_linear(const float* __restrict__ pooled,
                             const float* __restrict__ cnt,
                             const float* __restrict__ Wf,
                             const float* __restrict__ bf,
                             float* __restrict__ out) {
    __shared__ float sd[HC];
    int g = blockIdx.x, t = threadIdx.x;
    float c = cnt[g]; c = c > 1.0f ? c : 1.0f;
    sd[t] = (pooled[(size_t)g * HC + t] / c) * Wf[t];
    __syncthreads();
    for (int s = HC / 2; s > 0; s >>= 1) {
        if (t < s) sd[t] += sd[t + s];
        __syncthreads();
    }
    if (t == 0) out[g] = sd[0] + bf[0];
}

// ---------------- host-side orchestration ----------------
template <int K>
static void run_layer(const float* Xin,
                      const float* W, const float* as_w, const float* ad_w,
                      const float* bias, int do_relu,
                      const int* ei,
                      float* buf_h, float* buf_out,
                      float* a_s, float* a_d, float* m, float* denom,
                      hipStream_t stream) {
    // 1. GEMM: h = Xin * W^T   (3125 M-tiles x 8 N-strips / 8 waves)
    gemm_wmma<K><<<(NN / 16) * (HC / 32) / 8, 256, 0, stream>>>(Xin, W, buf_h);
    // 2. attention dots
    attn_dots<<<(NN * HH + 255) / 256, 256, 0, stream>>>(buf_h, as_w, ad_w, a_s, a_d);
    // 3. init m = -inf, denom = 0, out = 0
    fill_u32<<<(NN * HH + 255) / 256, 256, 0, stream>>>((unsigned int*)m, NN * HH, 0xFF800000u);
    fill_u32<<<(NN * HH + 255) / 256, 256, 0, stream>>>((unsigned int*)denom, NN * HH, 0u);
    fill_u32<<<(NN * HC + 255) / 256, 256, 0, stream>>>((unsigned int*)buf_out, NN * HC, 0u);
    // 4-6. edge passes (segment softmax + aggregate)
    edge_max<<<(ETOT + 255) / 256, 256, 0, stream>>>(ei, a_s, a_d, m);
    edge_expsum<<<(ETOT + 255) / 256, 256, 0, stream>>>(ei, a_s, a_d, m, denom);
    edge_scatter<<<(ETOT + 7) / 8, 256, 0, stream>>>(ei, a_s, a_d, m, denom, buf_h, buf_out);
    // 7. bias + relu
    finalize_bias<<<(NN * HC + 255) / 256, 256, 0, stream>>>(buf_out, bias, do_relu);
}

extern "C" void kernel_launch(void* const* d_in, const int* in_sizes, int n_in,
                              void* d_out, int out_size, void* d_ws, size_t ws_size,
                              hipStream_t stream) {
    const float* x    = (const float*)d_in[0];
    const int*   ei   = (const int*)d_in[1];
    const int*   batch= (const int*)d_in[2];
    const float* W1   = (const float*)d_in[3];
    const float* as1  = (const float*)d_in[4];
    const float* ad1  = (const float*)d_in[5];
    const float* b1   = (const float*)d_in[6];
    const float* W2   = (const float*)d_in[7];
    const float* as2  = (const float*)d_in[8];
    const float* ad2  = (const float*)d_in[9];
    const float* b2   = (const float*)d_in[10];
    const float* W3   = (const float*)d_in[11];
    const float* as3  = (const float*)d_in[12];
    const float* ad3  = (const float*)d_in[13];
    const float* b3   = (const float*)d_in[14];
    const float* Wf   = (const float*)d_in[15];
    const float* bf   = (const float*)d_in[16];
    float* out = (float*)d_out;

    // workspace carve-up (floats). Ping-pong big buffers stay L2-resident
    // (51.2 MB each, 192 MB L2 on MI455X).
    float* ws     = (float*)d_ws;
    float* buf_h  = ws;                       // [N, HC] transformed features
    float* buf_x  = buf_h + (size_t)NN * HC;  // [N, HC] layer in/out (ping-pong)
    float* a_s    = buf_x + (size_t)NN * HC;  // [N, H]
    float* a_d    = a_s + (size_t)NN * HH;
    float* m      = a_d + (size_t)NN * HH;
    float* denom  = m   + (size_t)NN * HH;
    float* pooled = denom + (size_t)NN * HH;  // [G, HC]
    float* cnt    = pooled + (size_t)GG * HC; // [G]

    // layer 1: x (K=128) -> buf_x
    run_layer<FIN>(x,     W1, as1, ad1, b1, 1, ei, buf_h, buf_x, a_s, a_d, m, denom, stream);
    // layer 2: buf_x (K=256) -> buf_x  (GEMM consumes buf_x before it is zeroed)
    run_layer<HC>(buf_x, W2, as2, ad2, b2, 1, ei, buf_h, buf_x, a_s, a_d, m, denom, stream);
    // layer 3: no relu
    run_layer<HC>(buf_x, W3, as3, ad3, b3, 0, ei, buf_h, buf_x, a_s, a_d, m, denom, stream);

    // global mean pool + final linear
    fill_u32<<<(GG * HC + 255) / 256, 256, 0, stream>>>((unsigned int*)pooled, GG * HC, 0u);
    fill_u32<<<1, 256, 0, stream>>>((unsigned int*)cnt, GG, 0u);
    pool_sum<<<NN, HC, 0, stream>>>(buf_x, batch, pooled, cnt);
    final_linear<<<GG, HC, 0, stream>>>(pooled, cnt, Wf, bf, out);
}
